// HeteroGCGRU_GAT_75642964017918
// MI455X (gfx1250) — matrix operand order repaired and checked by reading, compiled, not run on hardware
//
#include <hip/hip_runtime.h>
#include <hip/hip_bf16.h>
#include <math.h>

typedef __attribute__((ext_vector_type(16))) _Float16 v16h;
typedef __attribute__((ext_vector_type(8)))  _Float16 v8h;
typedef __attribute__((ext_vector_type(2)))  __fp16   fp16x2;  // cvt_pkrtz return type
typedef __attribute__((ext_vector_type(8)))  float    v8f;

static constexpr int kNA = 50000, kNB = 10000, kDIN = 32;
static constexpr int kEA2A = 800000, kEA2B = 400000, kEB2A = 400000;

// ---------------------------------------------------------------------------
// GEMM: C[M x 64] = A[M x K] (f32 row-major) * W[K x 64] (f32 row-major) + bias
// K in {32, 64}; M multiple of 16 (true: 50000, 10000).
// One wave computes a 16x64 tile with v_wmma_f32_16x16x32_f16.
// W staged in LDS transposed (sWt[n*K + k]) so each B fragment is two
// contiguous 16B LDS loads; A fragments are 4x float4 global loads + packed
// f32->f16 converts (v_cvt_pk_rtz_f16_f32).
// ---------------------------------------------------------------------------
__global__ void __launch_bounds__(256)
k_gemm_n64(const float* __restrict__ A, const float* __restrict__ W,
           const float* __restrict__ bias, float* __restrict__ C,
           int M, int K) {
  __shared__ _Float16 sWt[64 * 64];   // [n][k] fragment-order weights
  const int tid = threadIdx.x;
  for (int idx = tid; idx < K * 64; idx += 256) {
    const int n = idx & 63, k = idx >> 6;       // W[k*64+n]
    sWt[n * K + k] = (_Float16)W[idx];
  }
  __syncthreads();

  const int wave = tid >> 5;
  const int lane = tid & 31;
  const int row0 = (blockIdx.x * 8 + wave) * 16;
  if (row0 >= M) return;
  const int half = lane >> 4;   // 0: lanes 0-15, 1: lanes 16-31
  const int l16  = lane & 15;

  v8f acc[4] = {};
  const float* arow = A + (size_t)(row0 + l16) * K;
  for (int k0 = 0; k0 < K; k0 += 32) {
    // A fragment (16-bit A 16x32 layout): element e -> k = kb + (e/8)*16+(e%8)
    const int kb = k0 + half * 8;
    const float4* ap  = (const float4*)(arow + kb);
    const float4* ap2 = (const float4*)(arow + kb + 16);
    float4 f0 = ap[0], f1 = ap[1], f2 = ap2[0], f3 = ap2[1];
    union { v16h v; fp16x2 p[8]; } ua;
    ua.p[0] = __builtin_amdgcn_cvt_pkrtz(f0.x, f0.y);
    ua.p[1] = __builtin_amdgcn_cvt_pkrtz(f0.z, f0.w);
    ua.p[2] = __builtin_amdgcn_cvt_pkrtz(f1.x, f1.y);
    ua.p[3] = __builtin_amdgcn_cvt_pkrtz(f1.z, f1.w);
    ua.p[4] = __builtin_amdgcn_cvt_pkrtz(f2.x, f2.y);
    ua.p[5] = __builtin_amdgcn_cvt_pkrtz(f2.z, f2.w);
    ua.p[6] = __builtin_amdgcn_cvt_pkrtz(f3.x, f3.y);
    ua.p[7] = __builtin_amdgcn_cvt_pkrtz(f3.z, f3.w);
    const v16h a = ua.v;
    const int kc = k0 + half * 8;
#pragma unroll
    for (int nt = 0; nt < 4; ++nt) {
      const int n = nt * 16 + l16;
      const _Float16* bp = sWt + n * K + kc;   // two 16B runs (k, k+16), 16B aligned
      v8h b0 = *(const v8h*)bp;
      v8h b1 = *(const v8h*)(bp + 16);
      v16h b = __builtin_shufflevector(b0, b1, 0, 1, 2, 3, 4, 5, 6, 7,
                                       8, 9, 10, 11, 12, 13, 14, 15);
      acc[nt] = __builtin_amdgcn_wmma_f32_16x16x32_f16(
          false, a, false, b, (short)0, acc[nt], false, false);
    }
  }
  // C/D 16x16 f32 layout: lane holds col n = lane&15; VGPR r -> row r + 8*half
#pragma unroll
  for (int nt = 0; nt < 4; ++nt) {
    const int n = nt * 16 + l16;
    const float bv = bias ? bias[n] : 0.0f;
#pragma unroll
    for (int r = 0; r < 8; ++r) {
      const int m = row0 + r + half * 8;
      C[(size_t)m * 64 + n] = acc[nt][r] + bv;
    }
  }
}

// y[i] = dot(X[i, 0:64], v)
__global__ void k_matvec64(const float* __restrict__ X, const float* __restrict__ v,
                           float* __restrict__ y, int M) {
  int i = blockIdx.x * blockDim.x + threadIdx.x;
  if (i >= M) return;
  const float4* xr = (const float4*)(X + (size_t)i * 64);
  const float4* vv = (const float4*)v;
  float s = 0.f;
#pragma unroll
  for (int q = 0; q < 16; ++q) {
    float4 a = xr[q], b = vv[q];
    s += a.x * b.x + a.y * b.y + a.z * b.z + a.w * b.w;
  }
  y[i] = s;
}

// vd[k] = dot(Wd[k, 0:64], ad)   (collapses the w_d GEMM to a matvec)
__global__ void k_wv64(const float* __restrict__ Wd, const float* __restrict__ ad,
                       float* __restrict__ vd) {
  int k = threadIdx.x;  // 64 threads
  float s = 0.f;
  for (int j = 0; j < 64; ++j) s += Wd[k * 64 + j] * ad[j];
  vd[k] = s;
}

__global__ void k_fill_u32(unsigned* p, unsigned v, int n) {
  int i = blockIdx.x * blockDim.x + threadIdx.x;
  if (i < n) p[i] = v;
}
__global__ void k_fill_f32(float* p, float v, int n) {
  int i = blockIdx.x * blockDim.x + threadIdx.x;
  if (i < n) p[i] = v;
}
// conv[i*64+k] = b0[k] + (b1 ? b1[k] : 0)
__global__ void k_init_conv(float* __restrict__ conv, const float* __restrict__ b0,
                            const float* __restrict__ b1, int N) {
  int i = blockIdx.x * blockDim.x + threadIdx.x;
  if (i >= N * 64) return;
  int k = i & 63;
  conv[i] = b0[k] + (b1 ? b1[k] : 0.0f);
}

// monotonic float <-> uint keys for atomic segment-max
__device__ __forceinline__ unsigned fkey(float f) {
  unsigned u = __float_as_uint(f);
  return (u & 0x80000000u) ? ~u : (u | 0x80000000u);
}
__device__ __forceinline__ float fdec(unsigned k) {
  unsigned u = (k & 0x80000000u) ? (k & 0x7FFFFFFFu) : ~k;
  return __uint_as_float(u);
}

__global__ void k_edge_max(const int* __restrict__ src, const int* __restrict__ dst,
                           const float* __restrict__ als, const float* __restrict__ ald,
                           float* __restrict__ ebuf, unsigned* __restrict__ mkey, int E) {
  int i = blockIdx.x * blockDim.x + threadIdx.x;
  if (i >= E) return;
  if (i + 8192 < E) {  // stream the index arrays ahead (global_prefetch_b8)
    __builtin_prefetch(src + i + 8192, 0, 1);
    __builtin_prefetch(dst + i + 8192, 0, 1);
  }
  float e = als[src[i]] + ald[dst[i]];
  e = (e > 0.f) ? e : 0.2f * e;                 // leaky_relu(0.2)
  ebuf[i] = e;
  atomicMax(&mkey[dst[i]], fkey(e));
}

__global__ void k_edge_expsum(float* __restrict__ ebuf, const int* __restrict__ dst,
                              const unsigned* __restrict__ mkey,
                              float* __restrict__ den, int E) {
  int i = blockIdx.x * blockDim.x + threadIdx.x;
  if (i >= E) return;
  int d = dst[i];
  float ex = __expf(ebuf[i] - fdec(mkey[d]));
  ebuf[i] = ex;
  atomicAdd(&den[d], ex);
}

// conv[dst] += alpha * xs[src];  4 threads per edge, 16 features each
__global__ void k_edge_scatter(const float* __restrict__ ebuf, const int* __restrict__ src,
                               const int* __restrict__ dst, const float* __restrict__ den,
                               const float* __restrict__ xs, float* __restrict__ conv, int E) {
  int gid = blockIdx.x * blockDim.x + threadIdx.x;
  int i = gid >> 2;
  if (i >= E) return;
  int q = gid & 3;
  int d = dst[i], s = src[i];
  float alpha = ebuf[i] / (den[d] + 1e-16f);
  const float4* xr = (const float4*)(xs + (size_t)s * 64 + q * 16);
  float4 v0 = xr[0], v1 = xr[1], v2 = xr[2], v3 = xr[3];
  float* cr = conv + (size_t)d * 64 + q * 16;
  atomicAdd(&cr[0],  alpha * v0.x); atomicAdd(&cr[1],  alpha * v0.y);
  atomicAdd(&cr[2],  alpha * v0.z); atomicAdd(&cr[3],  alpha * v0.w);
  atomicAdd(&cr[4],  alpha * v1.x); atomicAdd(&cr[5],  alpha * v1.y);
  atomicAdd(&cr[6],  alpha * v1.z); atomicAdd(&cr[7],  alpha * v1.w);
  atomicAdd(&cr[8],  alpha * v2.x); atomicAdd(&cr[9],  alpha * v2.y);
  atomicAdd(&cr[10], alpha * v2.z); atomicAdd(&cr[11], alpha * v2.w);
  atomicAdd(&cr[12], alpha * v3.x); atomicAdd(&cr[13], alpha * v3.y);
  atomicAdd(&cr[14], alpha * v3.z); atomicAdd(&cr[15], alpha * v3.w);
}

__global__ void k_gate_sigmoid(float* __restrict__ g, const float* __restrict__ xg,
                               const float* __restrict__ conv, int n) {
  int i = blockIdx.x * blockDim.x + threadIdx.x;
  if (i < n) g[i] = 1.f / (1.f + __expf(-(xg[i] + conv[i])));
}
__global__ void k_mul_inplace(float* __restrict__ a, const float* __restrict__ b, int n) {
  int i = blockIdx.x * blockDim.x + threadIdx.x;
  if (i < n) a[i] *= b[i];
}
// out = (1-z)*tanh(xg+conv) + z*h
__global__ void k_final(float* __restrict__ out, const float* __restrict__ xg,
                        const float* __restrict__ conv, const float* __restrict__ z,
                        const float* __restrict__ h, int n) {
  int i = blockIdx.x * blockDim.x + threadIdx.x;
  if (i >= n) return;
  float nn = tanhf(xg[i] + conv[i]);
  out[i] = (1.f - z[i]) * nn + z[i] * h[i];
}

// ---------------------------------------------------------------------------
extern "C" void kernel_launch(void* const* d_in, const int* in_sizes, int n_in,
                              void* d_out, int out_size, void* d_ws, size_t ws_size,
                              hipStream_t stream) {
  (void)in_sizes; (void)n_in; (void)out_size; (void)ws_size;
  const float* x_a = (const float*)d_in[0];
  const float* x_b = (const float*)d_in[1];
  const float* h_a = (const float*)d_in[2];
  const float* h_b = (const float*)d_in[3];
  const float* Wx  = (const float*)d_in[4];   // [3,2,32,64]
  const float* bx  = (const float*)d_in[5];   // [3,2,64]
  const float* gws = (const float*)d_in[6];   // [3,3,64,64]
  const float* gwd = (const float*)d_in[7];   // [3,3,64,64]
  const float* gas = (const float*)d_in[8];   // [3,3,64]
  const float* gad = (const float*)d_in[9];   // [3,3,64]
  const float* gb  = (const float*)d_in[10];  // [3,3,64]
  const int* esrc[3] = {(const int*)d_in[11], (const int*)d_in[13], (const int*)d_in[15]};
  const int* edst[3] = {(const int*)d_in[12], (const int*)d_in[14], (const int*)d_in[16]};
  const int  ecnt[3] = {kEA2A, kEA2B, kEB2A};

  // workspace carve-up (256B-aligned slices)
  float* Wb = (float*)d_ws;
  size_t off = 0;
  auto alloc = [&](size_t n) { float* p = Wb + off; off += (n + 63) & ~(size_t)63; return p; };
  float* xg_a   = alloc((size_t)kNA * 64);
  float* xg_b   = alloc((size_t)kNB * 64);
  float* conv_a = alloc((size_t)kNA * 64);
  float* conv_b = alloc((size_t)kNB * 64);
  float* xs     = alloc((size_t)kNA * 64);   // reused per edge type
  float* z_a    = alloc((size_t)kNA * 64);
  float* z_b    = alloc((size_t)kNB * 64);
  float* r_a    = alloc((size_t)kNA * 64);   // becomes r*h in place
  float* r_b    = alloc((size_t)kNB * 64);
  float* als    = alloc(kNA);
  float* ald    = alloc(kNA);
  unsigned* mkey = (unsigned*)alloc(kNA);
  float* den    = alloc(kNA);
  float* vd     = alloc(64);
  float* ebuf   = alloc(kEA2A);              // reused per edge type

  const int TB = 256;
  auto g1 = [&](int n) { return dim3((unsigned)((n + TB - 1) / TB)); };

  // runs xg + hetero-conv for gate g with conv inputs (ha, hb); results in
  // xg_a/xg_b (incl. bx) and conv_a/conv_b (incl. GAT biases)
  auto run_conv = [&](int g, const float* ha, const float* hb) {
    k_gemm_n64<<<dim3((kNA + 127) / 128), TB, 0, stream>>>(
        x_a, Wx + ((size_t)g * 2 + 0) * kDIN * 64, bx + ((size_t)g * 2 + 0) * 64,
        xg_a, kNA, kDIN);
    k_gemm_n64<<<dim3((kNB + 127) / 128), TB, 0, stream>>>(
        x_b, Wx + ((size_t)g * 2 + 1) * kDIN * 64, bx + ((size_t)g * 2 + 1) * 64,
        xg_b, kNB, kDIN);
    k_init_conv<<<g1(kNA * 64), TB, 0, stream>>>(conv_a, gb + (g * 3 + 0) * 64,
                                                 gb + (g * 3 + 2) * 64, kNA);
    k_init_conv<<<g1(kNB * 64), TB, 0, stream>>>(conv_b, gb + (g * 3 + 1) * 64,
                                                 nullptr, kNB);
    for (int e = 0; e < 3; ++e) {            // 0:a2a 1:a2b 2:b2a
      const float* hs = (e == 2) ? hb : ha;
      const float* hd = (e == 1) ? hb : ha;
      const int Ns = (e == 2) ? kNB : kNA;
      const int Nd = (e == 1) ? kNB : kNA;
      float* convd = (e == 1) ? conv_b : conv_a;
      const int E = ecnt[e];
      const size_t w = (size_t)g * 3 + e;
      k_gemm_n64<<<dim3((Ns + 127) / 128), TB, 0, stream>>>(
          hs, gws + w * 64 * 64, nullptr, xs, Ns, 64);
      k_matvec64<<<g1(Ns), TB, 0, stream>>>(xs, gas + w * 64, als, Ns);
      k_wv64<<<1, 64, 0, stream>>>(gwd + w * 64 * 64, gad + w * 64, vd);
      k_matvec64<<<g1(Nd), TB, 0, stream>>>(hd, vd, ald, Nd);
      k_fill_u32<<<g1(Nd), TB, 0, stream>>>(mkey, 0u, Nd);
      k_fill_f32<<<g1(Nd), TB, 0, stream>>>(den, 0.f, Nd);
      k_edge_max<<<g1(E), TB, 0, stream>>>(esrc[e], edst[e], als, ald, ebuf, mkey, E);
      k_edge_expsum<<<g1(E), TB, 0, stream>>>(ebuf, edst[e], mkey, den, E);
      k_edge_scatter<<<g1(E * 4), TB, 0, stream>>>(ebuf, esrc[e], edst[e], den, xs,
                                                   convd, E);
    }
  };

  // z gate
  run_conv(0, h_a, h_b);
  k_gate_sigmoid<<<g1(kNA * 64), TB, 0, stream>>>(z_a, xg_a, conv_a, kNA * 64);
  k_gate_sigmoid<<<g1(kNB * 64), TB, 0, stream>>>(z_b, xg_b, conv_b, kNB * 64);
  // r gate
  run_conv(1, h_a, h_b);
  k_gate_sigmoid<<<g1(kNA * 64), TB, 0, stream>>>(r_a, xg_a, conv_a, kNA * 64);
  k_gate_sigmoid<<<g1(kNB * 64), TB, 0, stream>>>(r_b, xg_b, conv_b, kNB * 64);
  // hp = r * h (in place)
  k_mul_inplace<<<g1(kNA * 64), TB, 0, stream>>>(r_a, h_a, kNA * 64);
  k_mul_inplace<<<g1(kNB * 64), TB, 0, stream>>>(r_b, h_b, kNB * 64);
  // n gate + GRU blend straight into d_out ([new_a ; new_b])
  run_conv(2, r_a, r_b);
  float* out = (float*)d_out;
  k_final<<<g1(kNA * 64), TB, 0, stream>>>(out, xg_a, conv_a, z_a, h_a, kNA * 64);
  k_final<<<g1(kNB * 64), TB, 0, stream>>>(out + (size_t)kNA * 64, xg_b, conv_b,
                                           z_b, h_b, kNB * 64);
}
